// Mamba2Block_26113401160192
// MI455X (gfx1250) — matrix operand order repaired and checked by reading, not compiled, not run on hardware
//
#include <hip/hip_runtime.h>
#include <hip/hip_bf16.h>
#include <math.h>
#include <stdint.h>

#define SEQ     2048
#define DM      768
#define DI      1536
#define DSTATE  64
#define NH      24
#define HD      64
#define CONVCH  1664
#define DPROJ   3224
#define DPROJ_PAD 3232
#define FFH     2048
#define EPS_RMS 1.1920929e-07f
#define EPS_GATED 1e-05f
#define KC      32   // k-chunk per WMMA

typedef __bf16 bf16;
typedef bf16  v16bf __attribute__((ext_vector_type(16)));
typedef bf16  v8bf  __attribute__((ext_vector_type(8)));
typedef float v8f   __attribute__((ext_vector_type(8)));
typedef int   v4i   __attribute__((ext_vector_type(4)));

static __device__ __forceinline__ float silu_f(float v) {
    return v / (1.f + expf(-v));
}

// ---------------- CDNA5 async global->LDS copy (ASYNCcnt path) ----------------
#if defined(__AMDGCN__) && __has_builtin(__builtin_amdgcn_global_load_async_to_lds_b128)
#define HAVE_ASYNC_LDS 1
#else
#define HAVE_ASYNC_LDS 0
#endif

#if HAVE_ASYNC_LDS
static __device__ __forceinline__ void async_copy_b128(const bf16* g, bf16* l) {
    __builtin_amdgcn_global_load_async_to_lds_b128(
        (__attribute__((address_space(1))) v4i*)(g),
        (__attribute__((address_space(3))) v4i*)(l), 0, 0);
}
static __device__ __forceinline__ void wait_async0() {
#if __has_builtin(__builtin_amdgcn_s_wait_asynccnt)
    __builtin_amdgcn_s_wait_asynccnt(0);
#else
    asm volatile("s_wait_asynccnt 0x0" ::: "memory");
#endif
}
#endif

// ---------------------------------------------------------------- conversions
__global__ void k_f32_to_bf16(const float* __restrict__ s, bf16* __restrict__ d, int n) {
    int i = blockIdx.x * 256 + threadIdx.x;
    if (i < n) d[i] = (bf16)s[i];
}

__global__ void k_zero_bf16(bf16* __restrict__ d, int n) {
    int i = blockIdx.x * 256 + threadIdx.x;
    if (i < n) d[i] = (bf16)0.f;
}

// ---------------------------------------------------------------- RMSNorm -> bf16
__global__ void k_rmsnorm_bf16(const float* __restrict__ x, const float* __restrict__ w,
                               bf16* __restrict__ out, int D, float eps) {
    __shared__ float red[256];
    int row = blockIdx.x;
    const float* xr = x + (size_t)row * D;
    float s = 0.f;
    for (int i = threadIdx.x; i < D; i += 256) { float v = xr[i]; s += v * v; }
    red[threadIdx.x] = s;
    __syncthreads();
    for (int st = 128; st > 0; st >>= 1) {
        if ((int)threadIdx.x < st) red[threadIdx.x] += red[threadIdx.x + st];
        __syncthreads();
    }
    float scale = rsqrtf(red[0] / (float)D + eps);
    bf16* o = out + (size_t)row * D;
    for (int i = threadIdx.x; i < D; i += 256) o[i] = (bf16)(xr[i] * scale * w[i]);
}

// ---------------------------------------------------------------- WMMA GEMM
// C[M,N](f32) = A[M,K](bf16) @ W[N,K](bf16)^T (+resid).
// Block = 8 waves, tile 128(M) x 32(N). Weight tile (32 rows x 32 k = 2KB)
// double-buffered in LDS via global_load_async_to_lds_b128 (ASYNCcnt), shared
// by all waves; A fragments loaded direct (global_load_b128). Each wave owns
// one 16-row M tile and both 16-col N tiles => 2 WMMAs per chunk, A reused.
// A and B fragments use contiguous k = 16*hi + s (same order in both), not the ISA table order.
__global__ __launch_bounds__(256)
void k_gemm_bf16(const bf16* __restrict__ A, const bf16* __restrict__ W,
                 float* __restrict__ C, const float* __restrict__ resid,
                 int M, int N, int K, int ldc) {
    __shared__ bf16 sB[2][32][KC];   // [buf][weight row in tile][k]

    const int tid  = threadIdx.x;
    const int wave = tid >> 5;
    const int lane = tid & 31;
    const int l    = lane & 15;
    const int hi   = lane >> 4;
    const int n0   = blockIdx.x * 32;
    const int m0   = blockIdx.y * 128 + wave * 16;

    const bf16* arow = A + (size_t)(m0 + l) * K;

    // staging assignment: 128 loader threads, one b128 (8 bf16) each per chunk
    const int  r      = (tid & 127) >> 2;     // weight row in tile 0..31
    const int  seg    = tid & 3;              // 8-half segment 0..3
    const bool loader = tid < 128;
    const bf16* gsrc  = W + (size_t)(n0 + r) * K + seg * 8;

    v8f acc0, acc1;
#pragma unroll
    for (int i = 0; i < 8; ++i) { acc0[i] = 0.f; acc1[i] = 0.f; }

    const int nchunks = K / KC;

#if HAVE_ASYNC_LDS
    if (loader) async_copy_b128(gsrc, &sB[0][r][seg * 8]);
#endif

    for (int c = 0; c < nchunks; ++c) {
        const int k0  = c * KC;
        const int buf = c & 1;

#if HAVE_ASYNC_LDS
        wait_async0();           // this wave's async writes to LDS complete
        __syncthreads();         // all waves' tiles visible; prev compute done
        if (loader && (c + 1 < nchunks))
            async_copy_b128(gsrc + k0 + KC, &sB[buf ^ 1][r][seg * 8]);
#else
        __syncthreads();         // readers of this buffer (chunk c-2) are done
        if (loader)
            *(v8bf*)&sB[buf][r][seg * 8] = *(const v8bf*)(gsrc + k0);
        __syncthreads();
#endif

        union { v16bf v; v8bf h[2]; } a, b0, b1;
        const int ka = k0 + hi * 16;
        a.h[0] = *(const v8bf*)(arow + ka);
        a.h[1] = *(const v8bf*)(arow + ka + 8);
        if (k0 + 160 < K) __builtin_prefetch(arow + k0 + 128, 0, 1);

        b0.h[0] = *(const v8bf*)&sB[buf][l][hi * 16];
        b0.h[1] = *(const v8bf*)&sB[buf][l][hi * 16 + 8];
        b1.h[0] = *(const v8bf*)&sB[buf][16 + l][hi * 16];
        b1.h[1] = *(const v8bf*)&sB[buf][16 + l][hi * 16 + 8];

        acc0 = __builtin_amdgcn_wmma_f32_16x16x32_bf16(
                   false, a.v, false, b0.v, (short)0, acc0, false, false);
        acc1 = __builtin_amdgcn_wmma_f32_16x16x32_bf16(
                   false, a.v, false, b1.v, (short)0, acc1, false, false);
    }

    const int row0 = m0 + hi * 8;
    {
        const int col = n0 + l;
        if (col < N) {
            float* cp = C + (size_t)row0 * ldc + col;
            if (resid) {
                const float* rp = resid + (size_t)row0 * ldc + col;
#pragma unroll
                for (int q = 0; q < 8; ++q) cp[(size_t)q * ldc] = acc0[q] + rp[(size_t)q * ldc];
            } else {
#pragma unroll
                for (int q = 0; q < 8; ++q) cp[(size_t)q * ldc] = acc0[q];
            }
        }
    }
    {
        const int col = n0 + 16 + l;
        if (col < N) {
            float* cp = C + (size_t)row0 * ldc + col;
            if (resid) {
                const float* rp = resid + (size_t)row0 * ldc + col;
#pragma unroll
                for (int q = 0; q < 8; ++q) cp[(size_t)q * ldc] = acc1[q] + rp[(size_t)q * ldc];
            } else {
#pragma unroll
                for (int q = 0; q < 8; ++q) cp[(size_t)q * ldc] = acc1[q];
            }
        }
    }
}

// ---------------------------------------------------------------- conv(4)+SiLU, split
__global__ void k_conv_silu(const float* __restrict__ zx, const float* __restrict__ cw,
                            const float* __restrict__ cb, float* __restrict__ xh,
                            float* __restrict__ Bb, float* __restrict__ Cb) {
    int idx = blockIdx.x * 256 + threadIdx.x;
    if (idx >= SEQ * CONVCH) return;
    int t = idx / CONVCH, c = idx % CONVCH;
    float acc = cb[c];
#pragma unroll
    for (int k = 0; k < 4; ++k) {
        int tt = t + k - 3;
        if (tt >= 0) acc += zx[(size_t)tt * DPROJ + DI + c] * cw[c * 4 + k];
    }
    float s = silu_f(acc);
    if (c < DI)                 xh[(size_t)t * DI + c] = s;
    else if (c < DI + DSTATE)   Bb[t * DSTATE + (c - DI)] = s;
    else                        Cb[t * DSTATE + (c - DI - DSTATE)] = s;
}

// ---------------------------------------------------------------- dt = softplus(. + bias)
__global__ void k_dt(const float* __restrict__ zx, const float* __restrict__ dt_bias,
                     float* __restrict__ dt) {
    int idx = blockIdx.x * 256 + threadIdx.x;
    if (idx >= SEQ * NH) return;
    int t = idx / NH, h = idx % NH;
    float v = zx[(size_t)t * DPROJ + (2 * DI + 2 * DSTATE) + h] + dt_bias[h];
    dt[idx] = (v > 20.f) ? v : log1pf(expf(v));
}

// ---------------------------------------------------------------- SSM scan
// One block per head; 64 lanes each own one headdim row p with the full
// 64-wide state in registers. Double-buffered LDS for B/C => 1 barrier/step.
__global__ __launch_bounds__(64)
void k_scan(const float* __restrict__ dt, const float* __restrict__ xh,
            const float* __restrict__ Bb, const float* __restrict__ Cb,
            const float* __restrict__ A_log, const float* __restrict__ Dp,
            float* __restrict__ y) {
    __shared__ float sB[2][DSTATE];
    __shared__ float sC[2][DSTATE];
    const int h = blockIdx.x;
    const int p = threadIdx.x;          // 0..63
    const float Ah = -expf(A_log[h]);
    const float Dh = Dp[h];
    float s[DSTATE];
#pragma unroll
    for (int i = 0; i < DSTATE; ++i) s[i] = 0.f;

    for (int t = 0; t < SEQ; ++t) {
        const int buf = t & 1;
        sB[buf][p] = Bb[t * DSTATE + p];
        sC[buf][p] = Cb[t * DSTATE + p];
        const float dtv = dt[t * NH + h];
        const float xv  = xh[(size_t)t * DI + h * HD + p];
        __syncthreads();
        const float dA = expf(dtv * Ah);
        const float dx = dtv * xv;
        float yv = 0.f;
#pragma unroll
        for (int i = 0; i < DSTATE; ++i) {
            s[i] = s[i] * dA + dx * sB[buf][i];
            yv  += s[i] * sC[buf][i];
        }
        y[(size_t)t * DI + h * HD + p] = yv + Dh * xv;
    }
}

// ---------------------------------------------------------------- gated RMSNorm -> bf16
__global__ void k_gated_rmsnorm(const float* __restrict__ y, const float* __restrict__ zx,
                                const float* __restrict__ w, bf16* __restrict__ out) {
    __shared__ float ts[DI];
    __shared__ float red[256];
    int row = blockIdx.x;
    const float* yr = y + (size_t)row * DI;
    const float* zr = zx + (size_t)row * DPROJ;   // z = first DI cols of zxbcdt
    float s = 0.f;
    for (int i = threadIdx.x; i < DI; i += 256) {
        float t = yr[i] * silu_f(zr[i]);
        ts[i] = t;
        s += t * t;
    }
    red[threadIdx.x] = s;
    __syncthreads();
    for (int st = 128; st > 0; st >>= 1) {
        if ((int)threadIdx.x < st) red[threadIdx.x] += red[threadIdx.x + st];
        __syncthreads();
    }
    float scale = rsqrtf(red[0] / (float)DI + EPS_GATED);
    bf16* o = out + (size_t)row * DI;
    for (int i = threadIdx.x; i < DI; i += 256) o[i] = (bf16)(ts[i] * scale * w[i]);
}

// ---------------------------------------------------------------- FFN gating
__global__ void k_silu_mul_bf16(const float* __restrict__ g, const float* __restrict__ u,
                                bf16* __restrict__ out, int n) {
    int i = blockIdx.x * 256 + threadIdx.x;
    if (i < n) out[i] = (bf16)(silu_f(g[i]) * u[i]);
}

// ================================================================ launcher
extern "C" void kernel_launch(void* const* d_in, const int* in_sizes, int n_in,
                              void* d_out, int out_size, void* d_ws, size_t ws_size,
                              hipStream_t stream) {
    (void)in_sizes; (void)n_in; (void)out_size; (void)ws_size;
    const float* x          = (const float*)d_in[0];
    const float* norm1_w    = (const float*)d_in[1];
    const float* norm2_w    = (const float*)d_in[2];
    const float* in_proj_w  = (const float*)d_in[3];
    const float* conv_w     = (const float*)d_in[4];
    const float* conv_b     = (const float*)d_in[5];
    const float* dt_bias    = (const float*)d_in[6];
    const float* A_log      = (const float*)d_in[7];
    const float* Dp         = (const float*)d_in[8];
    const float* ssm_norm_w = (const float*)d_in[9];
    const float* out_proj_w = (const float*)d_in[10];
    const float* gate_w     = (const float*)d_in[11];
    const float* up_w       = (const float*)d_in[12];
    const float* down_w     = (const float*)d_in[13];
    float* out = (float*)d_out;

    uint8_t* base = (uint8_t*)d_ws;
    size_t off = 0;
    auto alloc = [&](size_t bytes) -> void* {
        void* p = base + off;
        off = (off + bytes + 255) & ~(size_t)255;
        return p;
    };

    bf16*  bw_in   = (bf16*)alloc((size_t)DPROJ_PAD * DM * 2);
    bf16*  bw_out  = (bf16*)alloc((size_t)DM * DI * 2);
    bf16*  bw_gate = (bf16*)alloc((size_t)FFH * DM * 2);
    bf16*  bw_up   = (bf16*)alloc((size_t)FFH * DM * 2);
    bf16*  bw_down = (bf16*)alloc((size_t)DM * FFH * 2);
    bf16*  h1      = (bf16*)alloc((size_t)SEQ * DM * 2);
    float* zx      = (float*)alloc((size_t)SEQ * DPROJ * 4);
    float* xh      = (float*)alloc((size_t)SEQ * DI * 4);
    float* Bb      = (float*)alloc((size_t)SEQ * DSTATE * 4);
    float* Cb      = (float*)alloc((size_t)SEQ * DSTATE * 4);
    float* dtb     = (float*)alloc((size_t)SEQ * NH * 4);
    float* ybuf    = (float*)alloc((size_t)SEQ * DI * 4);
    bf16*  yg      = (bf16*)alloc((size_t)SEQ * DI * 2);
    float* x1      = (float*)alloc((size_t)SEQ * DM * 4);
    bf16*  h2      = (bf16*)alloc((size_t)SEQ * DM * 2);
    float* gbuf    = (float*)alloc((size_t)SEQ * FFH * 4);
    float* ubuf    = (float*)alloc((size_t)SEQ * FFH * 4);
    bf16*  gact    = (bf16*)alloc((size_t)SEQ * FFH * 2);

    auto cdiv = [](int a, int b) { return (a + b - 1) / b; };

    // weights -> bf16 (pad in_proj N: 3224 -> 3232)
    k_f32_to_bf16<<<cdiv(DPROJ * DM, 256), 256, 0, stream>>>(in_proj_w, bw_in, DPROJ * DM);
    k_zero_bf16 <<<cdiv((DPROJ_PAD - DPROJ) * DM, 256), 256, 0, stream>>>(
        bw_in + (size_t)DPROJ * DM, (DPROJ_PAD - DPROJ) * DM);
    k_f32_to_bf16<<<cdiv(DM * DI, 256), 256, 0, stream>>>(out_proj_w, bw_out, DM * DI);
    k_f32_to_bf16<<<cdiv(FFH * DM, 256), 256, 0, stream>>>(gate_w, bw_gate, FFH * DM);
    k_f32_to_bf16<<<cdiv(FFH * DM, 256), 256, 0, stream>>>(up_w, bw_up, FFH * DM);
    k_f32_to_bf16<<<cdiv(DM * FFH, 256), 256, 0, stream>>>(down_w, bw_down, DM * FFH);

    // mixer
    k_rmsnorm_bf16<<<SEQ, 256, 0, stream>>>(x, norm1_w, h1, DM, EPS_RMS);
    k_gemm_bf16<<<dim3(DPROJ_PAD / 32, SEQ / 128), 256, 0, stream>>>(
        h1, bw_in, zx, nullptr, SEQ, DPROJ, DM, DPROJ);
    k_conv_silu<<<cdiv(SEQ * CONVCH, 256), 256, 0, stream>>>(zx, conv_w, conv_b, xh, Bb, Cb);
    k_dt<<<cdiv(SEQ * NH, 256), 256, 0, stream>>>(zx, dt_bias, dtb);
    k_scan<<<NH, 64, 0, stream>>>(dtb, xh, Bb, Cb, A_log, Dp, ybuf);
    k_gated_rmsnorm<<<SEQ, 256, 0, stream>>>(ybuf, zx, ssm_norm_w, yg);
    k_gemm_bf16<<<dim3(DM / 32, SEQ / 128), 256, 0, stream>>>(
        yg, bw_out, x1, x, SEQ, DM, DI, DM);          // x1 = x + mixer(x)

    // FFN
    k_rmsnorm_bf16<<<SEQ, 256, 0, stream>>>(x1, norm2_w, h2, DM, EPS_RMS);
    k_gemm_bf16<<<dim3(FFH / 32, SEQ / 128), 256, 0, stream>>>(
        h2, bw_gate, gbuf, nullptr, SEQ, FFH, DM, FFH);
    k_gemm_bf16<<<dim3(FFH / 32, SEQ / 128), 256, 0, stream>>>(
        h2, bw_up, ubuf, nullptr, SEQ, FFH, DM, FFH);
    k_silu_mul_bf16<<<cdiv(SEQ * FFH, 256), 256, 0, stream>>>(gbuf, ubuf, gact, SEQ * FFH);
    k_gemm_bf16<<<dim3(DM / 32, SEQ / 128), 256, 0, stream>>>(
        gact, bw_down, out, x1, SEQ, DM, FFH, DM);    // out = x1 + ffn(x1)
}
